// BNNUnet_41609643163681
// MI455X (gfx1250) — 8-bit matrix operand order repaired and checked by reading, compiled, not run on hardware
//
#include <hip/hip_runtime.h>
#include <hip/hip_bf16.h>
#include <stdint.h>


typedef int v8i __attribute__((ext_vector_type(8)));

#define WSTRIDE 20

__global__ __launch_bounds__(256) void prep_w_kernel(
    const float* __restrict__ w, int Cout, int Cin, int Npad,
    uint32_t* __restrict__ wp, float* __restrict__ sc)
{
  int tid = threadIdx.x;
  if (tid < Npad) {
    float s = 0.f;
    if (tid < Cout) {
      int n = Cin * 9;
      const float* wr = w + (size_t)tid * n;
      for (int i = 0; i < n; ++i) s += fabsf(wr[i]);
      s /= (float)n;
    }
    sc[tid] = s;
  }
  int total = 9 * Npad * WSTRIDE;
  for (int i = tid; i < total; i += 256) {
    int d   = i % WSTRIDE;
    int rem = i / WSTRIDE;
    int co  = rem % Npad;
    int t   = rem / Npad;
    uint32_t pk = 0u;
    if (co < Cout && d < 16) {
      for (int q = 0; q < 4; ++q) {
        int ci = d * 4 + q;
        int sb = 0;
        if (ci < Cin) {
          float f = w[((size_t)co * Cin + ci) * 9 + t];
          sb = (f > 0.f) ? 1 : ((f < 0.f) ? -1 : 0);
        }
        pk |= ((uint32_t)(uint8_t)(int8_t)sb) << (8 * q);
      }
    }
    wp[i] = pk;
  }
}

__global__ __launch_bounds__(256) void signpack_kernel(
    const float* __restrict__ in, uint32_t* __restrict__ sgn,
    int C, int Hout, int Wout, int mode)
{
  __shared__ float sX[48 * 64];
  int tid = threadIdx.x;
  int b = blockIdx.z;
  int y = blockIdx.y;
  int x0 = blockIdx.x * 64;

  if (mode == 2) {
    int Hin = Hout << 1, Win = Wout << 1;
    int yin = y << 1, xin0 = x0 << 1;
    for (int i = tid; i < C * 64; i += 256) {
      int c = i >> 6, xo = i & 63;
      const float* p = in + (((size_t)b * C + c) * Hin + yin) * Win + xin0 + (xo << 1);
      sX[c * 64 + xo] = 0.25f * (p[0] + p[1] + p[Win] + p[Win + 1]);
    }
  } else {
    int up2  = (mode == 1);
    int Hin  = up2 ? (Hout >> 1) : Hout;
    int Win  = up2 ? (Wout >> 1) : Wout;
    int yin  = up2 ? (y >> 1) : y;
    int WIN  = up2 ? 32 : 64;
    int xin0 = up2 ? (x0 >> 1) : x0;
    for (int i = tid; i < C * WIN; i += 256) {
      int c = i / WIN, xi = i - c * WIN;
      sX[c * 64 + xi] = in[(((size_t)b * C + c) * Hin + yin) * Win + xin0 + xi];
    }
  }
  __syncthreads();

  size_t pbase = ((size_t)b * Hout + y) * Wout + x0;
  for (int i = tid; i < 64 * 16; i += 256) {
    int d = i & 15;
    int x = i >> 4;
    int xi = (mode == 1) ? (x >> 1) : x;
    uint32_t pk = 0u;
    for (int q = 0; q < 4; ++q) {
      int c = d * 4 + q;
      int sb = 0;
      if (c < C) {
        float f = sX[c * 64 + xi];
        sb = (f > 0.f) ? 1 : ((f < 0.f) ? -1 : 0);
      }
      pk |= ((uint32_t)(uint8_t)(int8_t)sb) << (8 * q);
    }
    sgn[(pbase + x) * 16 + d] = pk;
  }
}

__global__ __launch_bounds__(256) void bconv_kernel(
    const uint32_t* __restrict__ sgn, const uint32_t* __restrict__ wp,
    const float* __restrict__ scales, const float* __restrict__ res,
    float* __restrict__ out, uint32_t* __restrict__ sgn_out,
    int Cout, int H, int W, int Npad, int accFlag, int emitUp)
{
  __shared__ uint32_t sA[18 * 18 * 17];
  __shared__ uint32_t sW[9 * 48 * WSTRIDE];
  __shared__ float    sO[8 * 16 * 17];
  __shared__ float    sS[48];

  int tid = threadIdx.x;
  int bx = blockIdx.x, by = blockIdx.y, bz = blockIdx.z;
  int gx0 = bx * 16 - 1, gy0 = by * 16 - 1;

  for (int i = tid; i < 324 * 4; i += 256) {
    int p = i >> 2, q = i & 3;
    int hy = p / 18, hx = p - hy * 18;
    int gy = gy0 + hy, gx = gx0 + hx;
    uint4 v = {0u, 0u, 0u, 0u};
    if (gy >= 0 && gy < H && gx >= 0 && gx < W) {
      const uint4* src = reinterpret_cast<const uint4*>(
          sgn + (((size_t)bz * H + gy) * W + gx) * 16) + q;
      v = *src;
    }
    int base = p * 17 + q * 4;
    sA[base + 0] = v.x; sA[base + 1] = v.y;
    sA[base + 2] = v.z; sA[base + 3] = v.w;
  }
  int nw = 9 * Npad * WSTRIDE;
  for (int i = tid; i < nw; i += 256) sW[i] = wp[i];
  if (tid < Npad) sS[tid] = scales[tid];
  __syncthreads();

  int lane = tid & 31, wv = tid >> 5;
  int half = lane >> 4, ml = lane & 15;
  int nT = Npad >> 4;
  int tiles = 16 * nT;

  for (int tix = wv; tix < tiles; tix += 8) {
    int row = tix / nT, nt = tix - row * nT;
    v8i acc = {0, 0, 0, 0, 0, 0, 0, 0};
#pragma unroll
    for (int t = 0; t < 9; ++t) {
      int r = t / 3, s = t - r * 3;
      int p  = (row + r) * 18 + (ml + s);
      int ab = p * 17 + half * 2;
      v8i a, b;
      a[0] = (int)sA[ab + 0];  a[1] = (int)sA[ab + 1];
      a[2] = (int)sA[ab + 4];  a[3] = (int)sA[ab + 5];
      a[4] = (int)sA[ab + 8];  a[5] = (int)sA[ab + 9];
      a[6] = (int)sA[ab + 12]; a[7] = (int)sA[ab + 13];
      int wb = (t * Npad + nt * 16 + ml) * WSTRIDE + half * 2;
      b[0] = (int)sW[wb + 0];  b[1] = (int)sW[wb + 1];
      b[2] = (int)sW[wb + 4];  b[3] = (int)sW[wb + 5];
      b[4] = (int)sW[wb + 8];  b[5] = (int)sW[wb + 9];
      b[6] = (int)sW[wb + 12]; b[7] = (int)sW[wb + 13];
      acc = __builtin_amdgcn_wmma_i32_16x16x64_iu8(true, a, true, b, acc,
                                                   false, false);
    }
    float sc = sS[nt * 16 + ml];
    int ob = wv * 272 + ml * 17 + half * 8;
#pragma unroll
    for (int j = 0; j < 8; ++j) sO[ob + j] = sc * (float)acc[j];
    int y = by * 16 + row;
#pragma unroll
    for (int it = 0; it < 8; ++it) {
      int n  = it * 2 + half;
      int co = nt * 16 + n;
      if (co < Cout) {
        int si = wv * 272 + n * 17 + ml;
        float v = sO[si];
        size_t oi = (((size_t)bz * Cout + co) * H + y) * W + bx * 16 + ml;
        if (res) v += res[oi];
        if (accFlag) v += out[oi];
        out[oi] = v;
        sO[si] = v;
      }
    }
    if (sgn_out) {
      uint32_t pk0 = 0u, pk1 = 0u;
#pragma unroll
      for (int q = 0; q < 4; ++q) {
        int n0 = (half * 2 + 0) * 4 + q;
        int n1 = (half * 2 + 1) * 4 + q;
        float f0 = sO[wv * 272 + n0 * 17 + ml];
        float f1 = sO[wv * 272 + n1 * 17 + ml];
        int s0 = (f0 > 0.f) ? 1 : ((f0 < 0.f) ? -1 : 0);
        int s1 = (f1 > 0.f) ? 1 : ((f1 < 0.f) ? -1 : 0);
        pk0 |= ((uint32_t)(uint8_t)(int8_t)s0) << (8 * q);
        pk1 |= ((uint32_t)(uint8_t)(int8_t)s1) << (8 * q);
      }
      uint2 v2; v2.x = pk0; v2.y = pk1;
      int dof = nt * 4 + half * 2;
      int gx = bx * 16 + ml;
      if (!emitUp) {
        size_t pix = ((size_t)bz * H + y) * W + gx;
        *reinterpret_cast<uint2*>(sgn_out + pix * 16 + dof) = v2;
        if (nt == 0 && half == 0) {
          uint4 z = {0u, 0u, 0u, 0u};
          for (int dz = nT * 4; dz < 16; dz += 4)
            *reinterpret_cast<uint4*>(sgn_out + pix * 16 + dz) = z;
        }
      } else {
        int H2 = H << 1, W2 = W << 1;
#pragma unroll
        for (int dy = 0; dy < 2; ++dy)
#pragma unroll
          for (int dx = 0; dx < 2; ++dx) {
            size_t pix = ((size_t)bz * H2 + 2 * y + dy) * W2 + 2 * gx + dx;
            *reinterpret_cast<uint2*>(sgn_out + pix * 16 + dof) = v2;
            if (nt == 0 && half == 0) {
              uint4 z = {0u, 0u, 0u, 0u};
              for (int dz = nT * 4; dz < 16; dz += 4)
                *reinterpret_cast<uint4*>(sgn_out + pix * 16 + dz) = z;
            }
          }
      }
    }
  }
}

__global__ __launch_bounds__(256) void add_kernel(
    float* __restrict__ out, const float* __restrict__ a)
{
  size_t i = (size_t)blockIdx.x * 256 + threadIdx.x;
  out[i] += a[i];
}

extern "C" void kernel_launch(void* const* d_in, const int* in_sizes, int n_in,
                              void* d_out, int out_size, void* d_ws, size_t ws_size,
                              hipStream_t stream)
{
  (void)in_sizes; (void)n_in; (void)out_size; (void)ws_size;
  const float* x = (const float*)d_in[0];
  float* out = (float*)d_out;
  char* ws = (char*)d_ws;
  size_t off = 0;
  auto alloc = [&](size_t bytes) -> void* {
    void* p = ws + off;
    off = (off + bytes + 255) & ~(size_t)255;
    return p;
  };
  const int WPS = 9 * 48 * WSTRIDE;
  uint32_t* WP  = (uint32_t*)alloc((size_t)20 * WPS * 4);
  float*    SC  = (float*)   alloc((size_t)20 * 48 * 4);
  uint32_t* S0  = (uint32_t*)alloc((size_t)4 * 512 * 512 * 64);
  uint32_t* S1  = (uint32_t*)alloc((size_t)4 * 512 * 512 * 64);
  float*    B1  = (float*)   alloc((size_t)4 * 24 * 512 * 512 * 4);
  float*    C2  = (float*)   alloc((size_t)4 * 36 * 256 * 256 * 4);
  float*    D2  = (float*)   alloc((size_t)4 * 36 * 256 * 256 * 4);
  float*    C3  = (float*)   alloc((size_t)4 * 48 * 128 * 128 * 4);

  struct LD { int widx, co, ci; };
  enum { L_ENC1 = 0, L_D12, L_E2_0, L_E2_1, L_E2_2, L_D23,
         L_E3_0, L_E3_1, L_E3_2, L_DC3_0, L_DC3_1, L_DC3_2,
         L_SK2, L_UP32, L_DC2_0, L_DC2_1, L_DC2_2, L_SK1, L_UP21, L_DC1 };
  const LD L[20] = {
    {1,24,24},{2,36,24},{3,36,36},{4,36,36},{5,36,36},{6,48,36},
    {7,48,48},{8,48,48},{9,48,48},{10,48,48},{11,48,48},{12,48,48},
    {13,36,36},{14,36,48},{15,36,36},{16,36,36},{17,36,36},
    {18,24,24},{19,24,36},{20,24,24}};

  for (int l = 0; l < 20; ++l) {
    int np = (L[l].co == 24) ? 32 : 48;
    prep_w_kernel<<<1, 256, 0, stream>>>((const float*)d_in[L[l].widx],
        L[l].co, L[l].ci, np, WP + (size_t)l * WPS, SC + (size_t)l * 48);
  }
  auto sp = [&](const float* src, int C, int Ho, int Wo, int mode, uint32_t* dst) {
    dim3 g(Wo / 64, Ho, 4);
    signpack_kernel<<<g, 256, 0, stream>>>(src, dst, C, Ho, Wo, mode);
  };
  auto cv = [&](int l, const uint32_t* sin, const float* res, float* o,
                int H, int W, int acc, uint32_t* sout, int up = 0) {
    int np = (L[l].co == 24) ? 32 : 48;
    dim3 g(W / 16, H / 16, 4);
    bconv_kernel<<<g, 256, 0, stream>>>(sin, WP + (size_t)l * WPS,
        SC + (size_t)l * 48, res, o, sout, L[l].co, H, W, np, acc, up);
  };

  sp(x, 24, 512, 512, 0, S0);
  cv(L_ENC1, S0, x, B1, 512, 512, 0, nullptr);
  sp(B1, 24, 256, 256, 2, S0);
  cv(L_D12,  S0, nullptr, C2, 256, 256, 0, S1);
  cv(L_E2_0, S1, C2, C2, 256, 256, 0, S0);
  cv(L_E2_1, S0, C2, C2, 256, 256, 0, S1);
  cv(L_E2_2, S1, C2, C2, 256, 256, 0, nullptr);
  sp(C2, 36, 128, 128, 2, S0);
  cv(L_D23,   S0, nullptr, C3, 128, 128, 0, S1);
  cv(L_E3_0,  S1, C3, C3, 128, 128, 0, S0);
  cv(L_E3_1,  S0, C3, C3, 128, 128, 0, S1);
  cv(L_E3_2,  S1, C3, C3, 128, 128, 0, S0);
  cv(L_DC3_0, S0, C3, C3, 128, 128, 0, S1);
  cv(L_DC3_1, S1, C3, C3, 128, 128, 0, S0);
  cv(L_DC3_2, S0, C3, C3, 128, 128, 0, S1, 1);
  cv(L_UP32, S1, nullptr, D2, 256, 256, 0, nullptr);
  sp(C2, 36, 256, 256, 0, S0);
  cv(L_SK2,   S0, C2, D2, 256, 256, 1, S1);
  cv(L_DC2_0, S1, D2, D2, 256, 256, 0, S0);
  cv(L_DC2_1, S0, D2, D2, 256, 256, 0, S1);
  cv(L_DC2_2, S1, D2, D2, 256, 256, 0, S0, 1);
  cv(L_UP21, S0, nullptr, out, 512, 512, 0, nullptr);
  sp(B1, 24, 512, 512, 0, S1);
  cv(L_SK1, S1, B1, out, 512, 512, 1, S0);
  cv(L_DC1, S0, out, out, 512, 512, 0, nullptr);
  add_kernel<<<(4 * 24 * 512 * 512) / 256, 256, 0, stream>>>(out, x);
}
